// MultiHeadAttention_35570919145890
// MI455X (gfx1250) — matrix operand order repaired and checked by reading, not compiled, not run on hardware
//
#include <hip/hip_runtime.h>

// ---------------------------------------------------------------------------
// MI455X (gfx1250) multi-head attention, bf16 WMMA pipeline, v2.
// B=2, T=2048, C=1024, H=16, D=64.
//  - ~69 GFLOP vs ~84 MB HBM traffic -> matrix-core bound; bf16 WMMA path.
//  - Projections: per-block async-LDS staging of the 64x1024 W super-tile
//    (128 KB of the 320 KB WGP LDS), 32x64 register tile per wave
//    (8 WMMA per 8 fragment loads).
//  - Attention: flash-style, 32 q-rows per wave, online softmax via 16-lane
//    shuffles, P repacked through per-wave LDS (in-order DS ops, no barrier).
// Workspace (bf16 elems): Xq,Xk,Xv [M*C] | Wq..Wo [C*C] | Q,K [B,H,T,D]
//                         | Vt [B,H,D,T] | Y [B,T,C]  == 64 MB total.
// ---------------------------------------------------------------------------

typedef __bf16 bf16_t;
typedef __attribute__((ext_vector_type(16))) __bf16 v16bf;
typedef __attribute__((ext_vector_type(8)))  __bf16 v8bf;
typedef __attribute__((ext_vector_type(4)))  __bf16 v4bf;
typedef __attribute__((ext_vector_type(8)))  float  v8f;

union FragAB { v16bf v; v8bf h[2]; };

__device__ __forceinline__ v8f wmma_bf16(v16bf a, v16bf b, v8f c) {
    // (neg_a, A, neg_b, B, c_mod, C, reuse_a, reuse_b)
    return __builtin_amdgcn_wmma_f32_16x16x32_bf16(false, a, false, b,
                                                   (short)0, c, false, false);
}

// Async global -> LDS copy of one 16B chunk per lane (ASYNCcnt-tracked).
__device__ __forceinline__ void async_copy_b128(const bf16_t* gsrc, bf16_t* ldst) {
    unsigned long long g = (unsigned long long)gsrc;
    unsigned l = (unsigned)(size_t)ldst;   // generic LDS addr: low 32 bits = LDS offset
    asm volatile("global_load_async_to_lds_b128 %0, %1, off"
                 :: "v"(l), "v"(g) : "memory");
}
__device__ __forceinline__ void wait_async0() {
    asm volatile("s_wait_asynccnt 0x0" ::: "memory");
}

// ---------------------------------------------------------------------------
// fp32 -> bf16 conversion (4-wide)
// ---------------------------------------------------------------------------
__global__ __launch_bounds__(256) void cvt_f32_to_bf16(
    const float* __restrict__ in, bf16_t* __restrict__ out, int n4) {
    int i = blockIdx.x * blockDim.x + threadIdx.x;
    if (i >= n4) return;
    float4 f = ((const float4*)in)[i];
    v4bf o;
    o[0] = (bf16_t)f.x; o[1] = (bf16_t)f.y;
    o[2] = (bf16_t)f.z; o[3] = (bf16_t)f.w;
    ((v4bf*)out)[i] = o;
}

// ---------------------------------------------------------------------------
// Projection: P = X @ W^T   (X:[4096,1024], W:[1024,1024] row-major).
// Block: 8 waves, 256x64 macro tile; W super-tile (64 rows x 1024 k = 128 KB)
// async-staged into LDS once, shared by all waves. Wave: 32x64 register tile.
// MODE 0: bf16 out [B,H,T,D]; 1: bf16 out [B,H,D,T]; 2: f32 out row-major.
// ---------------------------------------------------------------------------
template <int MODE>
__global__ __launch_bounds__(256) void proj_bf16_kernel(
    const bf16_t* __restrict__ X, const bf16_t* __restrict__ W,
    void* __restrict__ OutP) {
    const int C = 1024, T = 2048, H = 16, D = 64;
    extern __shared__ bf16_t wt[];               // [64][1024] bf16 = 128 KB

    int wid  = threadIdx.x >> 5;
    int lane = threadIdx.x & 31;
    int ln = lane & 15, hi = lane >> 4;
    int mb = blockIdx.x >> 4;                    // 16 macro row-blocks of 256
    int ns = blockIdx.x & 15;                    // 16 column super-tiles of 64
    int m0 = mb * 256 + wid * 32;
    int n0 = ns * 64;

    // ---- stage W rows [n0, n0+64) x all K into LDS (contiguous 128 KB) ----
    const bf16_t* wsrc = W + (size_t)n0 * C;
    for (int i = threadIdx.x; i < 8192; i += 256)       // 8192 x 16B chunks
        async_copy_b128(wsrc + (size_t)i * 8, wt + (size_t)i * 8);
    wait_async0();
    __syncthreads();

    v8f acc[2][4];
#pragma unroll
    for (int mt = 0; mt < 2; ++mt)
#pragma unroll
        for (int j = 0; j < 4; ++j) acc[mt][j] = (v8f){};

    const bf16_t* xrow0 = X + (size_t)(m0 + ln) * C;
    const bf16_t* xrow1 = X + (size_t)(m0 + 16 + ln) * C;
    const bf16_t* wl = wt + (size_t)ln * C + 16 * hi;   // per-lane LDS base

    for (int k = 0; k < C; k += 32) {
        FragAB a0, a1;
        // A-frag: lanes 0-15 K={k..k+15}; lanes 16-31 K={k+16..k+31}
        a0.h[0] = *(const v8bf*)(xrow0 + k + 16 * hi);
        a0.h[1] = *(const v8bf*)(xrow0 + k + 16 * hi + 8);
        a1.h[0] = *(const v8bf*)(xrow1 + k + 16 * hi);
        a1.h[1] = *(const v8bf*)(xrow1 + k + 16 * hi + 8);
#pragma unroll
        for (int j = 0; j < 4; ++j) {
            // B-frag from LDS: lane = column n (= W row 16j+ln), contiguous K
            v16bf b = *(const v16bf*)(wl + (size_t)(16 * j) * C + k);
            acc[0][j] = wmma_bf16(a0.v, b, acc[0][j]);
            acc[1][j] = wmma_bf16(a1.v, b, acc[1][j]);
        }
    }

    int b = m0 / T;
    int h = n0 / D;                              // one head per super-tile
#pragma unroll
    for (int mt = 0; mt < 2; ++mt) {
#pragma unroll
        for (int j = 0; j < 4; ++j) {
            int d = 16 * j + ln;
#pragma unroll
            for (int r = 0; r < 8; ++r) {
                int t = (m0 % T) + 16 * mt + r + 8 * hi;
                if (MODE == 0) {
                    ((bf16_t*)OutP)[((size_t)(b * H + h) * T + t) * D + d] =
                        (bf16_t)acc[mt][j][r];
                } else if (MODE == 1) {
                    ((bf16_t*)OutP)[((size_t)(b * H + h) * D + d) * T + t] =
                        (bf16_t)acc[mt][j][r];
                } else {
                    ((float*)OutP)[(size_t)(m0 + 16 * mt + r + 8 * hi) * C +
                                   n0 + 16 * j + ln] = acc[mt][j][r];
                }
            }
        }
    }
}

// ---------------------------------------------------------------------------
// Flash attention: Q,K [B,H,T,D] bf16, Vt [B,H,D,T] bf16 -> Y [B,T,C] bf16.
// 4 waves/block, each wave = 32 query rows; key blocks of 32; online softmax.
// 16 WMMA per key block against 8 global fragment loads.
// ---------------------------------------------------------------------------
__global__ __launch_bounds__(128) void attn_bf16_kernel(
    const bf16_t* __restrict__ Q, const bf16_t* __restrict__ K,
    const bf16_t* __restrict__ Vt, bf16_t* __restrict__ Y) {
    const int T = 2048, D = 64, H = 16, C = 1024;
    __shared__ __align__(16) bf16_t pbuf[4][2][16 * 32];   // 8 KB

    int wid  = threadIdx.x >> 5;
    int lane = threadIdx.x & 31;
    int ln = lane & 15, hi = lane >> 4;
    int bh = blockIdx.x >> 4;                    // 0..31 (B*H)
    int q0 = (blockIdx.x & 15) * 128 + wid * 32;

    const bf16_t* Qb = Q  + (size_t)bh * T * D;
    const bf16_t* Kb = K  + (size_t)bh * T * D;
    const bf16_t* Vb = Vt + (size_t)bh * D * T;

    // Q A-fragments: 2 m-subtiles x 2 K-chunks, kept in registers
    FragAB qa[2][2];
#pragma unroll
    for (int mt = 0; mt < 2; ++mt) {
        const bf16_t* qrow = Qb + (size_t)(q0 + 16 * mt + ln) * D;
#pragma unroll
        for (int c = 0; c < 2; ++c) {
            qa[mt][c].h[0] = *(const v8bf*)(qrow + 32 * c + 16 * hi);
            qa[mt][c].h[1] = *(const v8bf*)(qrow + 32 * c + 16 * hi + 8);
        }
    }

    v8f o[2][4];
    float mrow[2][8], lrow[2][8];
#pragma unroll
    for (int mt = 0; mt < 2; ++mt) {
#pragma unroll
        for (int j = 0; j < 4; ++j) o[mt][j] = (v8f){};
#pragma unroll
        for (int r = 0; r < 8; ++r) { mrow[mt][r] = -3.0e38f; lrow[mt][r] = 0.0f; }
    }

    const float scale = 0.125f;                  // 1/sqrt(64)

    for (int kb = 0; kb < T; kb += 32) {
        // ---- K fragments (shared by both m-subtiles) ----
        v16bf kf[2][2];
#pragma unroll
        for (int c = 0; c < 2; ++c)
#pragma unroll
            for (int nt = 0; nt < 2; ++nt)
                kf[c][nt] = *(const v16bf*)(Kb + (size_t)(kb + 16 * nt + ln) * D +
                                            32 * c + 16 * hi);
        // ---- scores ----
        v8f s[2][2];
#pragma unroll
        for (int mt = 0; mt < 2; ++mt)
#pragma unroll
            for (int nt = 0; nt < 2; ++nt) {
                v8f t = wmma_bf16(qa[mt][0].v, kf[0][nt], (v8f){});
                s[mt][nt] = wmma_bf16(qa[mt][1].v, kf[1][nt], t);
            }
        // ---- online softmax (acc layout: lane = col, VGPR r = row) ----
#pragma unroll
        for (int mt = 0; mt < 2; ++mt) {
            bf16_t* P = pbuf[wid][mt];
#pragma unroll
            for (int r = 0; r < 8; ++r) {
                float a = s[mt][0][r] * scale, bb = s[mt][1][r] * scale;
                float mx = fmaxf(a, bb);
                mx = fmaxf(mx, __shfl_xor(mx, 1, 32));
                mx = fmaxf(mx, __shfl_xor(mx, 2, 32));
                mx = fmaxf(mx, __shfl_xor(mx, 4, 32));
                mx = fmaxf(mx, __shfl_xor(mx, 8, 32));
                float mnew = fmaxf(mrow[mt][r], mx);
                float p0 = __expf(a - mnew);
                float p1 = __expf(bb - mnew);
                float corr = __expf(mrow[mt][r] - mnew);
                float ps = p0 + p1;
                ps += __shfl_xor(ps, 1, 32);
                ps += __shfl_xor(ps, 2, 32);
                ps += __shfl_xor(ps, 4, 32);
                ps += __shfl_xor(ps, 8, 32);
                lrow[mt][r] = lrow[mt][r] * corr + ps;
                mrow[mt][r] = mnew;
                o[mt][0][r] *= corr; o[mt][1][r] *= corr;
                o[mt][2][r] *= corr; o[mt][3][r] *= corr;
                P[(r + 8 * hi) * 32 + ln]      = (bf16_t)p0;
                P[(r + 8 * hi) * 32 + 16 + ln] = (bf16_t)p1;
            }
        }
        // ---- V fragments (shared by both m-subtiles) ----
        v16bf vf[4];
#pragma unroll
        for (int j = 0; j < 4; ++j)
            vf[j] = *(const v16bf*)(Vb + (size_t)(16 * j + ln) * T + kb + 16 * hi);
        // ---- O += P @ V  (P repacked through per-wave LDS, in-order DS) ----
#pragma unroll
        for (int mt = 0; mt < 2; ++mt) {
            FragAB pa;
            pa.h[0] = *(const v8bf*)(pbuf[wid][mt] + ln * 32 + 16 * hi);
            pa.h[1] = *(const v8bf*)(pbuf[wid][mt] + ln * 32 + 16 * hi + 8);
#pragma unroll
            for (int j = 0; j < 4; ++j)
                o[mt][j] = wmma_bf16(pa.v, vf[j], o[mt][j]);
        }
    }

    // ---- normalize + store to Y [B,T,C] ----
    int b = bh / H, h = bh % H;
#pragma unroll
    for (int mt = 0; mt < 2; ++mt)
#pragma unroll
        for (int r = 0; r < 8; ++r) {
            float inv = 1.0f / lrow[mt][r];
            int t = q0 + 16 * mt + r + 8 * hi;
#pragma unroll
            for (int j = 0; j < 4; ++j)
                Y[((size_t)b * T + t) * C + h * 64 + 16 * j + ln] =
                    (bf16_t)(o[mt][j][r] * inv);
        }
}

// ---------------------------------------------------------------------------
extern "C" void kernel_launch(void* const* d_in, const int* in_sizes, int n_in,
                              void* d_out, int out_size, void* d_ws, size_t ws_size,
                              hipStream_t stream) {
    const int B = 2, T = 2048, C = 1024;
    const size_t M  = (size_t)B * T;      // 4096
    const size_t MC = M * C;              // 4,194,304
    const size_t CC = (size_t)C * C;      // 1,048,576

    const float* q_in = (const float*)d_in[0];
    const float* k_in = (const float*)d_in[1];
    const float* v_in = (const float*)d_in[2];
    const float* Wq   = (const float*)d_in[3];
    const float* Wk   = (const float*)d_in[4];
    const float* Wv   = (const float*)d_in[5];
    const float* Wo   = (const float*)d_in[6];

    bf16_t* ws  = (bf16_t*)d_ws;
    bf16_t* Xq  = ws;                  // [M,C]
    bf16_t* Xk  = Xq  + MC;
    bf16_t* Xv  = Xk  + MC;
    bf16_t* Wqb = Xv  + MC;            // [C,C]
    bf16_t* Wkb = Wqb + CC;
    bf16_t* Wvb = Wkb + CC;
    bf16_t* Wob = Wvb + CC;
    bf16_t* Qp  = Wob + CC;            // [B,H,T,D]
    bf16_t* Kp  = Qp  + MC;            // [B,H,T,D]
    bf16_t* Vt  = Kp  + MC;            // [B,H,D,T]
    bf16_t* Yb  = Vt  + MC;            // [B,T,C]

    // fp32 -> bf16
    int mc4 = (int)(MC / 4), cc4 = (int)(CC / 4);
    cvt_f32_to_bf16<<<(mc4 + 255) / 256, 256, 0, stream>>>(q_in, Xq, mc4);
    cvt_f32_to_bf16<<<(mc4 + 255) / 256, 256, 0, stream>>>(k_in, Xk, mc4);
    cvt_f32_to_bf16<<<(mc4 + 255) / 256, 256, 0, stream>>>(v_in, Xv, mc4);
    cvt_f32_to_bf16<<<(cc4 + 255) / 256, 256, 0, stream>>>(Wq, Wqb, cc4);
    cvt_f32_to_bf16<<<(cc4 + 255) / 256, 256, 0, stream>>>(Wk, Wkb, cc4);
    cvt_f32_to_bf16<<<(cc4 + 255) / 256, 256, 0, stream>>>(Wv, Wvb, cc4);
    cvt_f32_to_bf16<<<(cc4 + 255) / 256, 256, 0, stream>>>(Wo, Wob, cc4);

    const size_t ldsW = 64 * 1024 * sizeof(bf16_t);      // 128 KB W super-tile

    // QKV projections: 256 blocks x 8 waves (32x64 tile per wave)
    proj_bf16_kernel<0><<<256, 256, ldsW, stream>>>(Xq, Wqb, Qp);
    proj_bf16_kernel<0><<<256, 256, ldsW, stream>>>(Xk, Wkb, Kp);
    proj_bf16_kernel<1><<<256, 256, ldsW, stream>>>(Xv, Wvb, Vt);  // V transposed

    // Attention: B*H * (T/128) = 512 blocks, 4 waves x 32 q-rows
    attn_bf16_kernel<<<512, 128, 0, stream>>>(Qp, Kp, Vt, Yb);

    // Output projection -> fp32 d_out
    proj_bf16_kernel<2><<<256, 256, ldsW, stream>>>(Yb, Wob, (float*)d_out);
}
